// ResidualVQ_88210038325338
// MI455X (gfx1250) — matrix operand order repaired and checked by reading, not compiled, not run on hardware
//
#include <hip/hip_runtime.h>

typedef __attribute__((ext_vector_type(16))) __bf16 v16bf;
typedef __attribute__((ext_vector_type(8)))  float  v8f;

#define NVQ   6
#define Bb    16
#define Cc    256
#define Hh    8
#define Ww    2048
#define Dd    8192
#define Tt    512
#define NTOK  8192
#define CD    64
#define CS    1024
#define NTILES 512

// ---------------- fragment helpers ----------------
// B-fragment (32x16 tile of a KxN matrix), 512 bf16 per tile, contiguous:
// lane l: n = nt*16 + (l&15); e<8 -> k = kt*32 + (l>>4)*8 + e ; e>=8 -> +16
__device__ __forceinline__ int bfrag_off(int k, int n, int NT) {
  int kt = k >> 5, kk = k & 31, nt = n >> 4, nr = n & 15;
  int lane = (((kk >> 3) & 1) << 4) | nr;
  int e = (kk & 7) | ((kk >> 4) << 3);
  return ((kt * NT + nt) << 9) + (lane << 4) + e;
}

__device__ __forceinline__ v16bf load_bfrag(const __bf16* base, int tile, int lane) {
  const v16bf* p = (const v16bf*)(base + ((long)tile << 9));
  return p[lane];
}

// A-fragment (16x32) from row-major bf16 with given pitch.
// lane l: m = l&15, khalf = l>>4; e<8 -> k=kt*32+khalf*8+e ; e>=8 -> +16
__device__ __forceinline__ v16bf load_afrag(const __bf16* base, int pitch, int m,
                                            int khalf, int kt) {
  const __bf16* p = base + m * pitch + (kt << 5) + (khalf << 3);
  v16bf a;
#pragma unroll
  for (int e = 0; e < 8; ++e) { a[e] = p[e]; a[8 + e] = p[16 + e]; }
  return a;
}

__device__ __forceinline__ v8f wmma_bf16(v16bf a, v16bf b, v8f c) {
  return __builtin_amdgcn_wmma_f32_16x16x32_bf16(false, a, false, b, (short)0, c,
                                                 false, false);
}

__device__ __forceinline__ unsigned int fkey(float v) {
  unsigned int b = __float_as_uint(v);
  return (b & 0x80000000u) ? ~b : (b | 0x80000000u);
}

// ---------------- prep: weights -> bf16 WMMA fragments ----------------
__global__ void k_prep_w(const float* __restrict__ wi, const float* __restrict__ wo,
                         const float* __restrict__ bo, __bf16* __restrict__ WiF,
                         __bf16* __restrict__ WoF, float* __restrict__ bo_sum) {
  long i = (long)blockIdx.x * blockDim.x + threadIdx.x;
  const long NW = (long)NVQ * Dd * CD;  // 3145728
  if (i < NW) {
    int s = (int)(i / (Dd * CD));
    int r = (int)(i % (Dd * CD));
    int k = r / CD, n = r % CD;
    WiF[(long)s * Dd * CD + bfrag_off(k, n, CD / 16)] = (__bf16)wi[i];
  } else if (i < 2 * NW) {
    long j = i - NW;
    int s = (int)(j / ((long)CD * Dd));
    long r = j % ((long)CD * Dd);
    int kk = (int)(r / Dd), n = (int)(r % Dd);
    WoF[bfrag_off(s * CD + kk, n, Dd / 16)] = (__bf16)wo[j];
  } else if (i < 2 * NW + Dd) {
    int d = (int)(i - 2 * NW);
    float s = 0.f;
    for (int j = 0; j < NVQ; ++j) s += bo[(long)j * Dd + d];
    bo_sum[d] = s;
  }
}

// ---------------- prep: normalized codebooks ----------------
__global__ void k_prep_cb(const float* __restrict__ cbk, __bf16* __restrict__ CbF,
                          __bf16* __restrict__ CbP) {
  int i = blockIdx.x * blockDim.x + threadIdx.x;
  if (i >= NVQ * CS) return;
  int s = i / CS, row = i % CS;
  const float* src = cbk + (long)i * CD;
  float ss = 0.f;
  for (int k = 0; k < CD; ++k) ss += src[k] * src[k];
  float sc = rsqrtf(ss + 1e-8f);
  for (int k = 0; k < CD; ++k) {
    __bf16 bv = (__bf16)(src[k] * sc);
    CbP[(long)i * CD + k] = bv;                                 // plain rows (gather)
    CbF[(long)s * CD * CS + bfrag_off(k, row, CS / 16)] = bv;   // B = cbn^T fragments
  }
}

// ---------------- prep: cross matrices -G_{j,s} = -(Wo_j @ Wi_s), bias fold h_s --------
__global__ void k_prep_g(const float* __restrict__ wi, const float* __restrict__ wo,
                         const float* __restrict__ bi, const float* __restrict__ bo,
                         __bf16* __restrict__ GF, float* __restrict__ hv) {
  int i = blockIdx.x * blockDim.x + threadIdx.x;
  const int NG = 15 * CD * CD;
  if (i < NG) {
    int pair = i / (CD * CD);
    int r = i % (CD * CD);
    int a = r / CD, nb = r % CD;
    int j = 0, s = 0, p = pair;
    for (j = 0;; ++j) { int cnt = 5 - j; if (p < cnt) { s = j + 1 + p; break; } p -= cnt; }
    const float* wor = wo + ((long)j * CD + a) * Dd;        // row of Wo_j
    const float* wic = wi + (long)s * Dd * CD + nb;         // column of Wi_s
    float acc = 0.f;
    for (int d = 0; d < Dd; ++d) acc += wor[d] * wic[(long)d * CD];
    GF[(long)(j * NVQ + s) * CD * CD + bfrag_off(a, nb, CD / 16)] = (__bf16)(-acc);
  } else if (i < NG + NVQ * CD) {
    int r = i - NG;
    int s = r / CD, n = r % CD;
    const float* wic = wi + (long)s * Dd * CD + n;
    float acc = bi[s * CD + n];
    for (int d = 0; d < Dd; ++d) {
      float pref = 0.f;
      for (int j = 0; j < s; ++j) pref += bo[(long)j * Dd + d];
      acc -= pref * wic[(long)d * CD];
    }
    hv[s * CD + n] = acc;  // = bi_s - (sum_{j<s} bo_j) @ Wi_s
  }
}

// ---------------- P_s = zf @ Wi_s + h_s for all 6 stages, one pass over z ---------
__global__ __launch_bounds__(256) void k_compP(const float* __restrict__ z,
                                               const __bf16* __restrict__ WiF,
                                               const float* __restrict__ hv,
                                               float* __restrict__ P) {
  __shared__ float Pl[8 * 16 * 64];  // per-wave partials, one stage at a time
  int tid = threadIdx.x, lane = tid & 31, wv = tid >> 5;
  int tile = blockIdx.x, tok0 = tile << 4;
  int m = lane & 15, khalf = lane >> 4;
  int g = tok0 + m;
  int bb = g >> 9, tt = g & 511;
  const float* zb = z + (long)bb * (Cc * Hh * Ww) + tt * 4;

  v8f acc[NVQ][4];
#pragma unroll
  for (int s = 0; s < NVQ; ++s)
#pragma unroll
    for (int nt = 0; nt < 4; ++nt)
#pragma unroll
      for (int r = 0; r < 8; ++r) acc[s][nt][r] = 0.f;

  int kt0 = wv * 32;  // 8 waves split K=8192 (256 k-tiles)
  for (int kt = kt0; kt < kt0 + 32; ++kt) {
    v16bf a;
#pragma unroll
    for (int half = 0; half < 2; ++half) {
      int kbase = (kt << 5) + (khalf << 3) + (half << 4);  // h==0 aligned chunk
      int o = kbase >> 11, rr = kbase & 2047, cc = rr >> 3;
      const float* zp = zb + (long)cc * (Hh * Ww) + o;
#pragma unroll
      for (int e = 0; e < 8; ++e) a[half * 8 + e] = (__bf16)zp[(long)e * Ww];
    }
#pragma unroll
    for (int s = 0; s < NVQ; ++s) {
      const __bf16* base = WiF + (long)s * Dd * CD;
#pragma unroll
      for (int nt = 0; nt < 4; ++nt) {
        v16bf b = load_bfrag(base, kt * 4 + nt, lane);
        acc[s][nt] = wmma_bf16(a, b, acc[s][nt]);
      }
    }
  }
  for (int s = 0; s < NVQ; ++s) {
#pragma unroll
    for (int nt = 0; nt < 4; ++nt)
#pragma unroll
      for (int r = 0; r < 8; ++r) {
        int mm = r + (khalf << 3), nn = (nt << 4) + m;
        Pl[wv * 1024 + mm * 64 + nn] = acc[s][nt][r];
      }
    __syncthreads();
    for (int i = tid; i < 1024; i += 256) {
      float v = hv[s * CD + (i & 63)];
#pragma unroll
      for (int w8 = 0; w8 < 8; ++w8) v += Pl[w8 * 1024 + i];
      P[((long)s * NTOK + tok0 + (i >> 6)) * CD + (i & 63)] = v;
    }
    __syncthreads();
  }
}

// ---------------- serial VQ over 6 stages, 16-token tiles in LDS ----------------
__global__ __launch_bounds__(256) void k_main(const float* __restrict__ P,
                                              const __bf16* __restrict__ GF,
                                              const __bf16* __restrict__ CbF,
                                              const __bf16* __restrict__ CbP,
                                              __bf16* __restrict__ Qbf,
                                              float* __restrict__ lossT) {
  __shared__ float pre[16 * 64];
  __shared__ __bf16 ebf[16 * 64];
  __shared__ __bf16 qs[NVQ][16 * 64];
  __shared__ unsigned long long best[16];
  __shared__ float red[256];
  int tid = threadIdx.x, lane = tid & 31, wv = tid >> 5;
  int tile = blockIdx.x, tok0 = tile << 4;
  int m = lane & 15, khalf = lane >> 4;
  float lossAcc = 0.f;

  for (int s = 0; s < NVQ; ++s) {
    // pre = P_s - sum_{j<s} q_j @ G_{j,s}   (waves 0..3 own one 16-col tile each)
    if (wv < 4) {
      int nt = wv;
      v8f c;
#pragma unroll
      for (int r = 0; r < 8; ++r) {
        int mm = r + (khalf << 3), nn = (nt << 4) + m;
        c[r] = P[((long)s * NTOK + tok0 + mm) * CD + nn];
      }
      for (int j = 0; j < s; ++j) {
        const __bf16* gbase = GF + (long)(j * NVQ + s) * CD * CD;
#pragma unroll
        for (int kt = 0; kt < 2; ++kt) {
          v16bf a = load_afrag(&qs[j][0], 64, m, khalf, kt);
          v16bf b = load_bfrag(gbase, kt * 4 + nt, lane);  // holds -G
          c = wmma_bf16(a, b, c);
        }
      }
#pragma unroll
      for (int r = 0; r < 8; ++r) {
        int mm = r + (khalf << 3), nn = (nt << 4) + m;
        pre[mm * 64 + nn] = c[r];
      }
    }
    if (tid < 16) best[tid] = 0ull;
    __syncthreads();
    // l2norm rows
    if (tid < 16) {
      float ssum = 0.f;
      for (int n = 0; n < 64; ++n) { float v = pre[tid * 64 + n]; ssum += v * v; }
      float sc = rsqrtf(ssum + 1e-8f);
      for (int n = 0; n < 64; ++n) {
        float v = pre[tid * 64 + n] * sc;
        pre[tid * 64 + n] = v;
        ebf[tid * 64 + n] = (__bf16)v;
      }
    }
    __syncthreads();
    // cosine sim vs 1024 codes + argmax (packed key, smaller index wins ties)
    const __bf16* cbase = CbF + (long)s * CD * CS;
    v16bf a0 = load_afrag(ebf, 64, m, khalf, 0);
    v16bf a1 = load_afrag(ebf, 64, m, khalf, 1);
    for (int ntl = 0; ntl < 8; ++ntl) {
      int nt = wv * 8 + ntl;
      v8f c;
#pragma unroll
      for (int r = 0; r < 8; ++r) c[r] = 0.f;
      c = wmma_bf16(a0, load_bfrag(cbase, nt, lane), c);
      c = wmma_bf16(a1, load_bfrag(cbase, 64 + nt, lane), c);
      int ncol = (nt << 4) + m;
#pragma unroll
      for (int r = 0; r < 8; ++r) {
        unsigned long long pk =
            ((unsigned long long)fkey(c[r]) << 32) | (unsigned)(1023 - ncol);
        atomicMax(&best[r + (khalf << 3)], pk);
      }
    }
    __syncthreads();
    // gather code, losses, persist Q
    {
      int mm = tid >> 4;
      int c0 = (tid & 15) << 2;
      int idx = 1023 - (int)((unsigned)best[mm] & 0x3FFu);
      const __bf16* crow = CbP + ((long)s * CS + idx) * CD;
#pragma unroll
      for (int jj = 0; jj < 4; ++jj) {
        int n = c0 + jj;
        __bf16 qv = crow[n];
        qs[s][mm * 64 + n] = qv;
        float d = pre[mm * 64 + n] - (float)qv;
        lossAcc += d * d;
        Qbf[((long)s * NTOK + tok0 + mm) * CD + n] = qv;
      }
    }
    __syncthreads();
  }
  red[tid] = lossAcc;
  __syncthreads();
  for (int off = 128; off > 0; off >>= 1) {
    if (tid < off) red[tid] += red[tid + off];
    __syncthreads();
  }
  if (tid == 0) lossT[tile] = red[0];
}

// ---------------- deterministic loss finalize (cm == cb in fwd) ----------------
__global__ void k_loss(const float* __restrict__ lossT, float* __restrict__ out_loss) {
  int b = threadIdx.x;
  if (b < 16) {
    float sum = 0.f;
    for (int i = 0; i < 32; ++i) sum += lossT[b * 32 + i];
    float v = sum * (1.0f / (Tt * CD));
    out_loss[b] = v;
    out_loss[16 + b] = v;
  }
}

// ---------------- zq = Q(16x384) @ Wo_stacked(384x8192) + bo_sum, coalesced out -----
__global__ __launch_bounds__(256) void k_final(const __bf16* __restrict__ Qbf,
                                               const __bf16* __restrict__ WoF,
                                               const float* __restrict__ bo_sum,
                                               float* __restrict__ out) {
  __shared__ __bf16 Ql[16 * 384];
  __shared__ __align__(16) float ob[8][16 * 64];
  int tid = threadIdx.x, lane = tid & 31, wv = tid >> 5;
  int tile = blockIdx.x, tok0 = tile << 4;
  int bb = tok0 >> 9, w0 = (tok0 & 511) << 2;
  for (int i = tid; i < 16 * 384; i += 256) {
    int mm = i / 384, kc = i % 384;
    int s = kc >> 6, kk = kc & 63;
    Ql[i] = Qbf[((long)s * NTOK + tok0 + mm) * CD + kk];
  }
  __syncthreads();
  int m = lane & 15, khalf = lane >> 4;
  float* myob = &ob[wv][0];
  float* outb = out + (long)bb * (Cc * Hh * Ww) + w0;
  for (int rbi = 0; rbi < 16; ++rbi) {
    int rb = wv + (rbi << 3);  // 0..127 blocks of r=c*8+h
    v8f c[4];
#pragma unroll
    for (int o = 0; o < 4; ++o) {
      float bias = bo_sum[(o << 11) + (rb << 4) + m];
#pragma unroll
      for (int r = 0; r < 8; ++r) c[o][r] = bias;
    }
#pragma unroll
    for (int kt = 0; kt < 12; ++kt) {
      v16bf a = load_afrag(Ql, 384, m, khalf, kt);
#pragma unroll
      for (int o = 0; o < 4; ++o) {
        v16bf b = load_bfrag(WoF, kt * 512 + (o << 7) + rb, lane);
        c[o] = wmma_bf16(a, b, c[o]);
      }
    }
#pragma unroll
    for (int o = 0; o < 4; ++o)
#pragma unroll
      for (int r = 0; r < 8; ++r) {
        int mrow = r + (khalf << 3);
        myob[m * 64 + (mrow << 2) + o] = c[o][r];  // row = nr, col = w-local
      }
    __syncthreads();
    for (int i2 = lane; i2 < 16 * 32; i2 += 32) {
      int nr = i2 >> 5, col2 = i2 & 31;
      float2 v = ((const float2*)myob)[nr * 32 + col2];
      int rg = (rb << 4) + nr;  // = c*8+h
      float* dst = outb + (long)rg * Ww + (col2 << 1);
      dst[0] = v.x;
      dst[1] = v.y;
    }
    __syncthreads();
  }
}

// ---------------- launcher ----------------
extern "C" void kernel_launch(void* const* d_in, const int* in_sizes, int n_in,
                              void* d_out, int out_size, void* d_ws, size_t ws_size,
                              hipStream_t stream) {
  const float* z  = (const float*)d_in[0];
  const float* wi = (const float*)d_in[1];
  const float* bi = (const float*)d_in[2];
  const float* wo = (const float*)d_in[3];
  const float* bo = (const float*)d_in[4];
  const float* cb = (const float*)d_in[5];
  float* out = (float*)d_out;
  float* out_loss = out + (long)Bb * Cc * Hh * Ww;

  char* w = (char*)d_ws;
  auto alloc = [&](size_t bytes) {
    char* p = w;
    w += (bytes + 255) & ~(size_t)255;
    return p;
  };
  __bf16* WiF  = (__bf16*)alloc((size_t)NVQ * Dd * CD * 2);
  __bf16* WoF  = (__bf16*)alloc((size_t)NVQ * CD * Dd * 2);
  __bf16* CbF  = (__bf16*)alloc((size_t)NVQ * CD * CS * 2);
  __bf16* CbP  = (__bf16*)alloc((size_t)NVQ * CS * CD * 2);
  __bf16* GF   = (__bf16*)alloc((size_t)36 * CD * CD * 2);
  __bf16* Qbf  = (__bf16*)alloc((size_t)NVQ * NTOK * CD * 2);
  float*  hv   = (float*)alloc((size_t)NVQ * CD * 4);
  float*  bo_s = (float*)alloc((size_t)Dd * 4);
  float*  P    = (float*)alloc((size_t)NVQ * NTOK * CD * 4);
  float*  lossT= (float*)alloc((size_t)NTILES * 4);

  long nprep = 2L * NVQ * Dd * CD + Dd;
  k_prep_w<<<(int)((nprep + 255) / 256), 256, 0, stream>>>(wi, wo, bo, WiF, WoF, bo_s);
  k_prep_cb<<<(NVQ * CS + 255) / 256, 256, 0, stream>>>(cb, CbF, CbP);
  int ng = 15 * CD * CD + NVQ * CD;
  k_prep_g<<<(ng + 255) / 256, 256, 0, stream>>>(wi, wo, bi, bo, GF, hv);
  k_compP<<<NTILES, 256, 0, stream>>>(z, WiF, hv, P);
  k_main<<<NTILES, 256, 0, stream>>>(P, GF, CbF, CbP, Qbf, lossT);
  k_loss<<<1, 32, 0, stream>>>(lossT, out_loss);
  k_final<<<NTILES, 256, 0, stream>>>(Qbf, WoF, bo_s, out);
}
